// Attention_85950885527709
// MI455X (gfx1250) — matrix operand order repaired and checked by reading, not compiled, not run on hardware
//
#include <hip/hip_runtime.h>
#include <hip/hip_bf16.h>

// ---------------------------------------------------------------------------
// EfficientViT-style attention block for MI455X (gfx1250, wave32, WMMA).
//
// ~19.4 GFLOP, <100 MB traffic when staged right -> hot loops must be
// b128-load + v_wmma dominated. All operands converted to f16 ONCE into
// WMMA-fragment-native layouts:
//   * GEMM B operands packed [kchunk][n][k%32] -> B frag = 2 contiguous b128
//   * qT/kT as (j, d) 16-half rows -> Q/K frags = 1-2 contiguous b128
//   * vh as (dv, j) rows (== v_img layout) -> V frag = 2 contiguous b128
// BN is pre-folded to per-channel (scale, bias). The 512 MB score tensor
// never leaves the WGP (flash attention, online softmax, P re-layout via a
// 1 KB LDS tile).
// ---------------------------------------------------------------------------

typedef __attribute__((ext_vector_type(16))) _Float16 v16h;
typedef __attribute__((ext_vector_type(8)))  _Float16 v8h;
typedef __attribute__((ext_vector_type(8)))  float    v8f;

union F16x16 { v16h v; v8h h[2]; };

#define DIM      256
#define NUM_HEAD 8
#define HEAD_DIM 32
#define KEY_DIM  16
#define QKV_CH   512
#define NPIX     1024          // 32*32
#define ATT_SCALE 0.25f        // KEY_DIM^-0.5 (folded into q storage)
#define BN_EPS   1e-5f

static __device__ __forceinline__ v8f wmma_f16(v16h a, v16h b, v8f c) {
  return __builtin_amdgcn_wmma_f32_16x16x32_f16(false, a, false, b,
                                                (short)0, c, false, false);
}

// ---------------------------------------------------------------------------
// Kernel 0a: plain f32 -> f16 conversion (weights).
// ---------------------------------------------------------------------------
__global__ void __launch_bounds__(256)
cvt_f16_kernel(const float* __restrict__ in, _Float16* __restrict__ out, int n)
{
  int i = blockIdx.x * 256 + threadIdx.x;
  if (i < n) out[i] = (_Float16)in[i];
}

// ---------------------------------------------------------------------------
// Kernel 0b: fold BN params to per-channel (scale, bias).
// ---------------------------------------------------------------------------
__global__ void __launch_bounds__(256)
bn_fold_kernel(const float* __restrict__ g, const float* __restrict__ b,
               const float* __restrict__ m, const float* __restrict__ v,
               float2* __restrict__ st, int n)
{
  int i = blockIdx.x * 256 + threadIdx.x;
  if (i < n) {
    float s = g[i] * rsqrtf(v[i] + BN_EPS);
    st[i] = make_float2(s, b[i] - m[i] * s);
  }
}

// ---------------------------------------------------------------------------
// Kernel 0c: pack a (B,256,1024) f32 tensor into the B-fragment-native f16
// layout: out[((b*8 + k/32)*1024 + n)*32 + k%32]. One thread per output row
// of 32 halfs: lane-coalesced reads, contiguous 64 B writes.
// grid = 512, block = 256  (16*8*1024 rows).
// ---------------------------------------------------------------------------
__global__ void __launch_bounds__(256)
pack_b_kernel(const float* __restrict__ in, _Float16* __restrict__ out)
{
  const int idx = blockIdx.x * 256 + threadIdx.x;   // (b, kc, n)
  const int n  = idx & (NPIX - 1);
  const int kc = (idx >> 10) & 7;
  const int b  = idx >> 13;
  const float* src = in + ((size_t)b * 256 + kc * 32) * NPIX + n;
  _Float16* dst = out + (size_t)idx * 32;
#pragma unroll
  for (int i = 0; i < 32; ++i) dst[i] = (_Float16)src[(size_t)i * NPIX];
}

// ---------------------------------------------------------------------------
// Kernel 1/4: GEMM + folded BN, all-f16, one wave per 16x64 output strip.
// Per K-chunk: A frag (2 b128) + 4 B frags (8 b128) loaded first, then 4
// back-to-back WMMAs, so load waits stagger instead of serializing.
//   qkv mode (Of32==null): scatter BN result into WMMA-native f16 tensors.
//   proj mode: plain f32 store to Of32.
// grid = (Ngroups=16, Mtiles=M/16, B=16), block = 32.
// ---------------------------------------------------------------------------
__global__ void __launch_bounds__(32)
gemm_bn_kernel(const _Float16* __restrict__ W16,  // M x 256 row-major f16
               const _Float16* __restrict__ Xp,   // packed B operand (see 0c)
               const float2* __restrict__ st,     // folded BN per channel
               float* __restrict__ Of32,          // proj mode output (or null)
               _Float16* __restrict__ qT, _Float16* __restrict__ kT,
               _Float16* __restrict__ vh)         // qkv mode outputs (or null)
{
  const int lane = threadIdx.x;
  const int n0 = blockIdx.x * 64;
  const int m0 = blockIdx.y * 16;
  const int b  = blockIdx.z;
  const _Float16* xb = Xp + (size_t)b * 256 * NPIX;   // 8 kchunks * 1024 * 32

  const int mA = lane & 15;      // A-frag: lane owns row M
  const int hi = lane >> 4;      // half-wave id
  const int nB = lane & 15;      // B/C/D-frag: lane owns column N

  v8f acc[4] = {{}, {}, {}, {}};
  for (int kc = 0; kc < 8; ++kc) {
    if (kc + 1 < 8)              // gfx1250 global_prefetch of next K-chunk
      __builtin_prefetch(xb + ((size_t)(kc + 1) * NPIX + n0 + nB) * 32, 0, 1);

    F16x16 a;                    // A 16x32: K = hi*16 + idx
    a.h[0] = *(const v8h*)(W16 + (m0 + mA) * 256 + kc * 32 + (hi << 4));
    a.h[1] = *(const v8h*)(W16 + (m0 + mA) * 256 + kc * 32 + (hi << 4) + 8);

    F16x16 bf[4];                // B 32x16: lane nB, K = hi*16..hi*16+15
#pragma unroll
    for (int nt = 0; nt < 4; ++nt) {
      const _Float16* xr =
          xb + ((size_t)kc * NPIX + n0 + (nt << 4) + nB) * 32 + (hi << 4);
      bf[nt].h[0] = *(const v8h*)(xr);
      bf[nt].h[1] = *(const v8h*)(xr + 8);
    }
#pragma unroll
    for (int nt = 0; nt < 4; ++nt)
      acc[nt] = wmma_f16(a.v, bf[nt].v, acc[nt]);
  }

  // Folded-BN epilogue: y = acc*s + t; D row m = r + 8*hi, col n0 + nt*16 + nB.
#pragma unroll
  for (int r = 0; r < 8; ++r) {
    const int c = m0 + r + (hi << 3);
    const float2 f = st[c];
    if (qT) {                                  // qkv mode: f16 scatter stores
      const int h = c >> 6, r64 = c & 63;
#pragma unroll
      for (int nt = 0; nt < 4; ++nt) {
        const int n = n0 + (nt << 4) + nB;
        const float y = acc[nt][r] * f.x + f.y;
        if (r64 < KEY_DIM) {
          qT[((size_t)(b * NUM_HEAD + h) * NPIX + n) * KEY_DIM + r64] =
              (_Float16)(y * ATT_SCALE);
        } else if (r64 < 2 * KEY_DIM) {
          kT[((size_t)(b * NUM_HEAD + h) * NPIX + n) * KEY_DIM + (r64 - KEY_DIM)] =
              (_Float16)y;
        } else {
          vh[((size_t)b * DIM + h * HEAD_DIM + (r64 - 2 * KEY_DIM)) * NPIX + n] =
              (_Float16)y;
        }
      }
    } else {                                   // proj mode: f32 output
#pragma unroll
      for (int nt = 0; nt < 4; ++nt)
        Of32[((size_t)b * DIM + c) * NPIX + n0 + (nt << 4) + nB] =
            acc[nt][r] * f.x + f.y;
    }
  }
}

// ---------------------------------------------------------------------------
// Kernel 2: fused flash attention, one wave per (b, h, 16-query tile).
// All fragments are contiguous f16 b128 loads; zero conversions in the loop.
// grid = (64, 8, 16), block = 32.
// ---------------------------------------------------------------------------
__global__ void __launch_bounds__(32)
attn_kernel(const _Float16* __restrict__ qT, const _Float16* __restrict__ kT,
            const _Float16* __restrict__ vh, float* __restrict__ out)
{
  __shared__ _Float16 pTile[16 * 32];
  const int lane = threadIdx.x;
  const int i0 = blockIdx.x * 16;
  const int h  = blockIdx.y;
  const int b  = blockIdx.z;
  const _Float16* qTb = qT + (size_t)(b * NUM_HEAD + h) * NPIX * KEY_DIM;
  const _Float16* kTb = kT + (size_t)(b * NUM_HEAD + h) * NPIX * KEY_DIM;
  const _Float16* vhb = vh + (size_t)(b * DIM + h * HEAD_DIM) * NPIX;

  const int mA = lane & 15;
  const int hi = lane >> 4;
  const int nB = lane & 15;
  const v8h zer = {};

  // Q A-fragment (16 queries x 32 K, K>=16 zero-pad), loaded once.
  F16x16 qa;
  qa.h[0] = hi ? zer : *(const v8h*)(qTb + (i0 + mA) * KEY_DIM);
  qa.h[1] = hi ? zer : *(const v8h*)(qTb + (i0 + mA) * KEY_DIM + 8);

  float rmax[8], rsum[8];
  v8f o0 = {}, o1 = {};
#pragma unroll
  for (int r = 0; r < 8; ++r) { rmax[r] = -1e30f; rsum[r] = 0.0f; }

  for (int j0 = 0; j0 < NPIX; j0 += 32) {
    // ---- scores: two 16x16 tiles over keys j0..j0+31 ----
    F16x16 kb0, kb1;             // B 32x16: hi=0 lanes carry d=0..15, hi=1 pad
    if (hi == 0) {
      const _Float16* r0 = kTb + (j0 + nB) * KEY_DIM;
      const _Float16* r1 = kTb + (j0 + 16 + nB) * KEY_DIM;
      kb0.h[0] = *(const v8h*)(r0); kb0.h[1] = *(const v8h*)(r0 + 8);
      kb1.h[0] = *(const v8h*)(r1); kb1.h[1] = *(const v8h*)(r1 + 8);
    } else {
      kb0.h[0] = zer; kb0.h[1] = zer; kb1.h[0] = zer; kb1.h[1] = zer;
    }
    v8f zero = {};
    v8f s0 = wmma_f16(qa.v, kb0.v, zero);
    v8f s1 = wmma_f16(qa.v, kb1.v, zero);

    // ---- online softmax over the 32 new columns (scale pre-folded) ----
#pragma unroll
    for (int r = 0; r < 8; ++r) {
      float v0 = s0[r], v1 = s1[r];
      float cm = fmaxf(v0, v1);
      cm = fmaxf(cm, __shfl_xor(cm, 1, 32)); // masks <16 stay in half-wave
      cm = fmaxf(cm, __shfl_xor(cm, 2, 32));
      cm = fmaxf(cm, __shfl_xor(cm, 4, 32));
      cm = fmaxf(cm, __shfl_xor(cm, 8, 32));
      float nm    = fmaxf(rmax[r], cm);
      float alpha = __expf(rmax[r] - nm);
      float p0 = __expf(v0 - nm);
      float p1 = __expf(v1 - nm);
      float cs = p0 + p1;
      cs += __shfl_xor(cs, 1, 32);
      cs += __shfl_xor(cs, 2, 32);
      cs += __shfl_xor(cs, 4, 32);
      cs += __shfl_xor(cs, 8, 32);
      rsum[r] = rsum[r] * alpha + cs;
      rmax[r] = nm;
      o0[r] *= alpha;
      o1[r] *= alpha;
      s0[r] = p0;
      s1[r] = p1;
    }

    // ---- P: C-layout -> LDS -> A-layout (16 x 32 f16) ----
#pragma unroll
    for (int r = 0; r < 8; ++r) {
      int m = r + (hi << 3);
      pTile[m * 32 + nB]      = (_Float16)s0[r];
      pTile[m * 32 + 16 + nB] = (_Float16)s1[r];
    }
    __syncthreads();
    F16x16 pa;                                 // ds_load_b128 x2
    pa.h[0] = *(const v8h*)(&pTile[mA * 32 + (hi << 4)]);
    pa.h[1] = *(const v8h*)(&pTile[mA * 32 + (hi << 4) + 8]);
    __syncthreads();

    // ---- O += P (16i x 32j) @ V^T (32j x 16dv), two dv halves ----
    F16x16 vb0, vb1;
    const _Float16* v0r = vhb + (size_t)nB * NPIX + j0 + (hi << 4);
    const _Float16* v1r = vhb + (size_t)(16 + nB) * NPIX + j0 + (hi << 4);
    vb0.h[0] = *(const v8h*)(v0r); vb0.h[1] = *(const v8h*)(v0r + 8);
    vb1.h[0] = *(const v8h*)(v1r); vb1.h[1] = *(const v8h*)(v1r + 8);
    o0 = wmma_f16(pa.v, vb0.v, o0);
    o1 = wmma_f16(pa.v, vb1.v, o1);
  }

  // ---- normalize and store f32: out[b, h*32 + dv, i] ----
  float* ob = out + ((size_t)b * DIM + (size_t)h * HEAD_DIM) * NPIX;
#pragma unroll
  for (int r = 0; r < 8; ++r) {
    int m = r + (hi << 3);
    float inv = 1.0f / rsum[r];
    ob[(size_t)nB * NPIX + i0 + m]        = o0[r] * inv;
    ob[(size_t)(16 + nB) * NPIX + i0 + m] = o1[r] * inv;
  }
}

// ---------------------------------------------------------------------------
// Kernel 3: depthwise 3x3 conv on V (f16 vh) + folded BN, accumulated in
// place (f32) onto the attention output. grid = (B*256), block = 256.
// ---------------------------------------------------------------------------
__global__ void __launch_bounds__(256)
pe_add_kernel(const _Float16* __restrict__ vh, const float* __restrict__ pw,
              const float2* __restrict__ st, float* __restrict__ z)
{
  const int bc = blockIdx.x;
  const int b = bc >> 8, c = bc & 255;
  const _Float16* vimg = vh + ((size_t)b * DIM + c) * NPIX;
  float w[9];
#pragma unroll
  for (int i = 0; i < 9; ++i) w[i] = pw[c * 9 + i];
  const float2 f = st[c];
  float* zb = z + ((size_t)b * DIM + c) * NPIX;

  for (int p = threadIdx.x; p < NPIX; p += 256) {
    int y = p >> 5, x = p & 31;
    float acc = 0.0f;
#pragma unroll
    for (int dy = -1; dy <= 1; ++dy)
#pragma unroll
      for (int dx = -1; dx <= 1; ++dx) {
        int yy = y + dy, xx = x + dx;
        if (yy >= 0 && yy < 32 && xx >= 0 && xx < 32)
          acc += (float)vimg[yy * 32 + xx] * w[(dy + 1) * 3 + (dx + 1)];
      }
    zb[p] += acc * f.x + f.y;
  }
}

// ---------------------------------------------------------------------------
// Launch. Workspace (~51 MB):
//   xp packed f16 (8 MB) | w16q/w16p f16 (0.4 MB) | qT,kT (4 MB ea)
//   vh (8 MB) | zp packed f16 (8 MB) | zbuf f32 (16 MB) | folded BN (8 KB)
// ---------------------------------------------------------------------------
extern "C" void kernel_launch(void* const* d_in, const int* in_sizes, int n_in,
                              void* d_out, int out_size, void* d_ws, size_t ws_size,
                              hipStream_t stream) {
  const float* x      = (const float*)d_in[0];
  const float* qkv_w  = (const float*)d_in[1];
  const float* qkv_g  = (const float*)d_in[2];
  const float* qkv_b  = (const float*)d_in[3];
  const float* qkv_m  = (const float*)d_in[4];
  const float* qkv_v  = (const float*)d_in[5];
  const float* pe_w   = (const float*)d_in[6];
  const float* pe_g   = (const float*)d_in[7];
  const float* pe_b   = (const float*)d_in[8];
  const float* pe_m   = (const float*)d_in[9];
  const float* pe_v   = (const float*)d_in[10];
  const float* proj_w = (const float*)d_in[11];
  const float* proj_g = (const float*)d_in[12];
  const float* proj_b = (const float*)d_in[13];
  const float* proj_m = (const float*)d_in[14];
  const float* proj_v = (const float*)d_in[15];

  _Float16* xp   = (_Float16*)d_ws;             // 4,194,304 halfs
  _Float16* w16q = xp   + (size_t)4194304;      //   131,072
  _Float16* w16p = w16q + (size_t)131072;       //    65,536
  _Float16* qT   = w16p + (size_t)65536;        // 2,097,152
  _Float16* kT   = qT   + (size_t)2097152;      // 2,097,152
  _Float16* vh   = kT   + (size_t)2097152;      // 4,194,304
  _Float16* zp   = vh   + (size_t)4194304;      // 4,194,304
  float*    zbuf = (float*)(zp + (size_t)4194304); // 4,194,304 f32
  float2*   stq  = (float2*)(zbuf + (size_t)4194304); // 512
  float2*   stp  = stq + 512;                   // 256
  float2*   stpe = stp + 256;                   // 256
  float*    y    = (float*)d_out;

  // 0) one-shot conversions / packs / BN folds
  pack_b_kernel<<<dim3(512), 256, 0, stream>>>(x, xp);
  cvt_f16_kernel<<<dim3(512), 256, 0, stream>>>(qkv_w, w16q, 512 * 256);
  cvt_f16_kernel<<<dim3(256), 256, 0, stream>>>(proj_w, w16p, 256 * 256);
  bn_fold_kernel<<<dim3(2), 256, 0, stream>>>(qkv_g, qkv_b, qkv_m, qkv_v, stq, 512);
  bn_fold_kernel<<<dim3(1), 256, 0, stream>>>(proj_g, proj_b, proj_m, proj_v, stp, 256);
  bn_fold_kernel<<<dim3(1), 256, 0, stream>>>(pe_g, pe_b, pe_m, pe_v, stpe, 256);

  // 1) QKV GEMM + BN -> qT (pre-scaled), kT, vh (all f16, WMMA-native)
  gemm_bn_kernel<<<dim3(16, 32, 16), 32, 0, stream>>>(
      w16q, xp, stq, nullptr, qT, kT, vh);
  // 2) fused softmax attention -> zbuf (f32)
  attn_kernel<<<dim3(64, 8, 16), 32, 0, stream>>>(qT, kT, vh, zbuf);
  // 3) zbuf += BN(depthwise3x3(vh))   (f32, in place)
  pe_add_kernel<<<dim3(16 * 256), 256, 0, stream>>>(vh, pe_w, stpe, zbuf);
  // 3b) pack zbuf into proj GEMM B-operand layout
  pack_b_kernel<<<dim3(512), 256, 0, stream>>>(zbuf, zp);
  // 4) y = BN(proj_w @ zp)  (f32 output)
  gemm_bn_kernel<<<dim3(16, 16, 16), 32, 0, stream>>>(
      w16p, zp, stp, y, nullptr, nullptr, nullptr);
}
